// GraphAutoEncoder_50757923504169
// MI455X (gfx1250) — matrix operand order repaired and checked by reading, not compiled, not run on hardware
//
#include <hip/hip_runtime.h>
#include <hip/hip_bf16.h>
#include <math.h>

// ---------------------------------------------------------------------------
// GATv2 graph autoencoder for MI455X (gfx1250, wave32).
// Activations/weights are converted to bf16 once per layer; the dense
// transforms run a fused dual-B WMMA GEMM (shared A fragment, two
// V_WMMA_F32_16X16X32_BF16 per K-step -> Wl and Wr outputs in one pass).
// Edge attention / segment softmax / scatter use wave-per-edge kernels with
// float atomics (gathers stay L2-resident: biggest xl is 41MB < 192MB L2).
// ---------------------------------------------------------------------------

typedef __attribute__((ext_vector_type(16))) __bf16 v16bf;
typedef __attribute__((ext_vector_type(8)))  float  v8f;

#define NN     20000
#define EEDGE  320000
#define D_IN   512
#define HID1   256
#define HID2   64
#define EDIM   8
#define NEDGE  (EEDGE + NN)     // edges + self loops
#define NEG_SLOPE 0.2f

// fp32 -> bf16 bulk conversion (round via clang __bf16 conversion).
__global__ void cvt_bf16(const float* __restrict__ in, __bf16* __restrict__ out,
                         size_t n) {
    size_t i = (size_t)blockIdx.x * blockDim.x + threadIdx.x;
    if (i < n) out[i] = (__bf16)in[i];
}

// ---------------------------------------------------------------------------
// Fused dual GEMM: C1 = A*W1^T + b1, C2 = A*W2^T + b2 (A shared).
// A[M x K] bf16, W1/W2[Nout x K] bf16, C f32. M%16==0, K%32==0, Nout%16==0.
// Block = (32,4): 4 waves, each owns one 16x16 tile pair (same N, 4 M tiles).
// Two v_wmma_f32_16x16x32_bf16 per K-step on a shared A fragment.
// ---------------------------------------------------------------------------
__global__ void wmma_gemm_dual(const __bf16* __restrict__ A,
                               const __bf16* __restrict__ W1,
                               const __bf16* __restrict__ W2,
                               const float* __restrict__ b1,
                               const float* __restrict__ b2,
                               float* __restrict__ C1,
                               float* __restrict__ C2,
                               int M, int K, int Nout) {
    const int lane  = threadIdx.x;                       // 0..31
    const int tileM = (blockIdx.y * blockDim.y + threadIdx.y) * 16;
    const int tileN = blockIdx.x * 16;
    if (tileM >= M) return;                              // wave-uniform exit

    const int half = lane >> 4;                          // 0: lanes 0-15, 1: 16-31
    const int l15  = lane & 15;

    const __bf16* __restrict__ arow  = A  + (size_t)(tileM + l15) * K;
    const __bf16* __restrict__ w1row = W1 + (size_t)(tileN + l15) * K;
    const __bf16* __restrict__ w2row = W2 + (size_t)(tileN + l15) * K;

    v8f acc1 = {};
    v8f acc2 = {};
    for (int k0 = 0; k0 < K; k0 += 32) {
        v16bf a, b1v, b2v;
        // A 16x32 bf16 layout: lane half selects +8 K-offset; VGPRs 4..7 add +16.
        #pragma unroll
        for (int r = 0; r < 8; ++r) {
            const int k = k0 + ((r & 4) << 2) + (half << 3) + ((r & 3) << 1);
            a[2 * r]     = arow[k];
            a[2 * r + 1] = arow[k + 1];
        }
        // B 32x16 bf16 layout: same K order as A (lane half selects +8 K-offset; VGPRs 4..7 add +16).
        #pragma unroll
        for (int r = 0; r < 8; ++r) {
            const int k = k0 + ((r & 4) << 2) + (half << 3) + ((r & 3) << 1);
            b1v[2 * r]     = w1row[k];
            b1v[2 * r + 1] = w1row[k + 1];
            b2v[2 * r]     = w2row[k];
            b2v[2 * r + 1] = w2row[k + 1];
        }
        acc1 = __builtin_amdgcn_wmma_f32_16x16x32_bf16(
            false, a, false, b1v, (short)0, acc1, false, false);
        acc2 = __builtin_amdgcn_wmma_f32_16x16x32_bf16(
            false, a, false, b2v, (short)0, acc2, false, false);
    }

    // C/D layout: VGPR v -> row (half?8:0)+v, col = l15.
    const int   mOff = half ? 8 : 0;
    const int   nCol = tileN + l15;
    const float bv1  = b1 ? b1[nCol] : 0.0f;
    const float bv2  = b2 ? b2[nCol] : 0.0f;
    #pragma unroll
    for (int v = 0; v < 8; ++v) {
        const size_t idx = (size_t)(tileM + mOff + v) * Nout + nCol;
        C1[idx] = acc1[v] + bv1;
        C2[idx] = acc2[v] + bv2;
    }
}

// ---------------------------------------------------------------------------
// Edge-feature mean (self-loop fill_value='mean'): thread-local accum + atomics.
// ---------------------------------------------------------------------------
__global__ void ef_mean_accum(const float* __restrict__ ef, float* __restrict__ acc) {
    float s[EDIM] = {};
    for (int e = blockIdx.x * blockDim.x + threadIdx.x; e < EEDGE;
         e += gridDim.x * blockDim.x) {
        #pragma unroll
        for (int k = 0; k < EDIM; ++k) s[k] += ef[(size_t)e * EDIM + k];
    }
    #pragma unroll
    for (int k = 0; k < EDIM; ++k) atomicAdd(&acc[k], s[k]);
}

__global__ void scale_vec(float* __restrict__ v, float s, int n) {
    int i = blockIdx.x * blockDim.x + threadIdx.x;
    if (i < n) v[i] *= s;
}

__global__ void init_neg_inf(int* __restrict__ p, int n) {
    int i = blockIdx.x * blockDim.x + threadIdx.x;
    if (i < n) p[i] = 0xFF800000;    // -inf bit pattern
}

__global__ void init_bias(float* __restrict__ out, const float* __restrict__ b,
                          int dout, int total) {
    int i = blockIdx.x * blockDim.x + threadIdx.x;
    if (i < total) out[i] = b[i % dout];
}

// ---------------------------------------------------------------------------
// Per-edge attention score: wave per edge, lanes stride over dout.
// score[e] = sum_j leaky( xl[src,j] + xr[dst,j] + (ef[e]·We[j,:]) ) * att[j]
// ---------------------------------------------------------------------------
__global__ void edge_score(const float* __restrict__ xl, const float* __restrict__ xr,
                           const long long* __restrict__ ei,
                           const float* __restrict__ ef, const float* __restrict__ efm,
                           const float* __restrict__ We, const float* __restrict__ att,
                           float* __restrict__ score, int dout) {
    const int e    = blockIdx.x * (blockDim.x >> 5) + (threadIdx.x >> 5);
    const int lane = threadIdx.x & 31;
    if (e >= NEDGE) return;

    long long s, d;
    float f[EDIM];
    if (e < EEDGE) {
        s = ei[e]; d = ei[EEDGE + e];
        #pragma unroll
        for (int k = 0; k < EDIM; ++k) f[k] = ef[(size_t)e * EDIM + k];
    } else {
        s = d = (long long)(e - EEDGE);
        #pragma unroll
        for (int k = 0; k < EDIM; ++k) f[k] = efm[k];
    }

    const float* __restrict__ xls = xl + (size_t)s * dout;
    const float* __restrict__ xrd = xr + (size_t)d * dout;
    float sum = 0.0f;
    for (int j = lane; j < dout; j += 32) {
        float eej = 0.0f;
        #pragma unroll
        for (int k = 0; k < EDIM; ++k) eej += f[k] * We[(size_t)j * EDIM + k];
        const float m  = xls[j] + xrd[j] + eej;
        const float lm = (m > 0.0f) ? m : NEG_SLOPE * m;
        sum += lm * att[j];
    }
    #pragma unroll
    for (int off = 16; off > 0; off >>= 1) sum += __shfl_down(sum, off, 32);
    if (lane == 0) score[e] = sum;
}

// Segment max over destinations (float-as-int monotonic atomic trick).
__global__ void seg_max(const float* __restrict__ score, const long long* __restrict__ ei,
                        int* __restrict__ smax) {
    int e = blockIdx.x * blockDim.x + threadIdx.x;
    if (e >= NEDGE) return;
    const long long d = (e < EEDGE) ? ei[EEDGE + e] : (long long)(e - EEDGE);
    const int iv = __float_as_int(score[e]);
    if (iv >= 0) atomicMax(&smax[d], iv);
    else         atomicMin((unsigned int*)&smax[d], (unsigned int)iv);
}

// ex[e] = exp(score - smax[dst]);  denom[dst] += ex[e]
__global__ void exp_denom(const float* __restrict__ score, const long long* __restrict__ ei,
                          const int* __restrict__ smax, float* __restrict__ ex,
                          float* __restrict__ denom) {
    int e = blockIdx.x * blockDim.x + threadIdx.x;
    if (e >= NEDGE) return;
    const long long d = (e < EEDGE) ? ei[EEDGE + e] : (long long)(e - EEDGE);
    const float v = expf(score[e] - __int_as_float(smax[d]));
    ex[e] = v;
    atomicAdd(&denom[d], v);
}

// out[dst,:] += (ex[e]/denom[dst]) * xl[src,:]  — wave per edge.
__global__ void scatter_out(const float* __restrict__ xl, const long long* __restrict__ ei,
                            const float* __restrict__ ex, const float* __restrict__ denom,
                            float* __restrict__ out, int dout) {
    const int e    = blockIdx.x * (blockDim.x >> 5) + (threadIdx.x >> 5);
    const int lane = threadIdx.x & 31;
    if (e >= NEDGE) return;
    long long s, d;
    if (e < EEDGE) { s = ei[e]; d = ei[EEDGE + e]; }
    else           { s = d = (long long)(e - EEDGE); }
    const float alpha = ex[e] / denom[d];
    const float* __restrict__ xls = xl + (size_t)s * dout;
    float* __restrict__ od = out + (size_t)d * dout;
    for (int j = lane; j < dout; j += 32) atomicAdd(&od[j], alpha * xls[j]);
}

// z = mu + 0.5*exp(logvar)*eps
__global__ void reparam(const float* __restrict__ mu, const float* __restrict__ lv,
                        const float* __restrict__ eps, float* __restrict__ z, int n) {
    int i = blockIdx.x * blockDim.x + threadIdx.x;
    if (i < n) z[i] = mu[i] + 0.5f * expf(lv[i]) * eps[i];
}

// ---------------------------------------------------------------------------
// Host-side: one full GATv2 layer (bf16 staging + fused dual WMMA GEMM).
// ---------------------------------------------------------------------------
static void run_gat_layer(const float* hin, int din, int dout,
                          const float* Wl, const float* Wr, const float* We,
                          const float* att, const float* b,
                          const long long* ei, const float* ef, const float* efm,
                          __bf16* abf, __bf16* w1bf, __bf16* w2bf,
                          float* xl, float* xr, float* score, float* ex,
                          int* smax, float* denom, float* out, hipStream_t stream) {
    const size_t na = (size_t)NN * din;
    const size_t nw = (size_t)dout * din;
    cvt_bf16<<<(na + 255) / 256, 256, 0, stream>>>(hin, abf, na);
    cvt_bf16<<<(nw + 255) / 256, 256, 0, stream>>>(Wl, w1bf, nw);
    cvt_bf16<<<(nw + 255) / 256, 256, 0, stream>>>(Wr, w2bf, nw);

    dim3 gblk(32, 4);
    dim3 ggrid(dout / 16, (NN / 16 + 3) / 4);
    wmma_gemm_dual<<<ggrid, gblk, 0, stream>>>(abf, w1bf, w2bf, nullptr, nullptr,
                                               xl, xr, NN, din, dout);

    init_neg_inf<<<(NN + 255) / 256, 256, 0, stream>>>(smax, NN);
    hipMemsetAsync(denom, 0, NN * sizeof(float), stream);

    edge_score<<<(NEDGE + 7) / 8, 256, 0, stream>>>(xl, xr, ei, ef, efm, We, att, score, dout);
    seg_max<<<(NEDGE + 255) / 256, 256, 0, stream>>>(score, ei, smax);
    exp_denom<<<(NEDGE + 255) / 256, 256, 0, stream>>>(score, ei, smax, ex, denom);

    init_bias<<<((size_t)NN * dout + 255) / 256, 256, 0, stream>>>(out, b, dout, NN * dout);
    scatter_out<<<(NEDGE + 7) / 8, 256, 0, stream>>>(xl, ei, ex, denom, out, dout);
}

extern "C" void kernel_launch(void* const* d_in, const int* in_sizes, int n_in,
                              void* d_out, int out_size, void* d_ws, size_t ws_size,
                              hipStream_t stream) {
    // Top-level inputs in setup_inputs() insertion order.
    const float*     x   = (const float*)d_in[0];       // [N, 512]
    const long long* ei  = (const long long*)d_in[1];   // [2, E] int64
    const float*     ef  = (const float*)d_in[2];       // [E, 8]
    const float*     eps = (const float*)d_in[3];       // [N, 64]
    // params flattened with JAX sorted-key dict order:
    //   dec[0]{We,Wl,Wr,att,b}, dec[1]{...}, enc[0]{...}, enc[1]{...},
    //   lv_W, lv_b, mu_W, mu_b
    const float* dWe0 = (const float*)d_in[4];   // [256,8]
    const float* dWl0 = (const float*)d_in[5];   // [256,64]
    const float* dWr0 = (const float*)d_in[6];   // [256,64]
    const float* datt0= (const float*)d_in[7];   // [256]
    const float* db0  = (const float*)d_in[8];   // [256]
    const float* dWe1 = (const float*)d_in[9];   // [512,8]
    const float* dWl1 = (const float*)d_in[10];  // [512,256]
    const float* dWr1 = (const float*)d_in[11];  // [512,256]
    const float* datt1= (const float*)d_in[12];  // [512]
    const float* db1  = (const float*)d_in[13];  // [512]
    const float* eWe0 = (const float*)d_in[14];  // [256,8]
    const float* eWl0 = (const float*)d_in[15];  // [256,512]
    const float* eWr0 = (const float*)d_in[16];  // [256,512]
    const float* eatt0= (const float*)d_in[17];  // [256]
    const float* eb0  = (const float*)d_in[18];  // [256]
    const float* eWe1 = (const float*)d_in[19];  // [64,8]
    const float* eWl1 = (const float*)d_in[20];  // [64,256]
    const float* eWr1 = (const float*)d_in[21];  // [64,256]
    const float* eatt1= (const float*)d_in[22];  // [64]
    const float* eb1  = (const float*)d_in[23];  // [64]
    const float* lvW  = (const float*)d_in[24];  // [64,64]
    const float* lvb  = (const float*)d_in[25];  // [64]
    const float* muW  = (const float*)d_in[26];  // [64,64]
    const float* mub  = (const float*)d_in[27];  // [64]

    // Workspace carving (float-granular offsets).
    float* ws = (float*)d_ws;
    size_t o = 0;
    float* xl    = ws + o; o += (size_t)NN * 512;
    float* xr    = ws + o; o += (size_t)NN * 512;
    float* h1    = ws + o; o += (size_t)NN * 256;   // enc1 out; reused as dec0 out
    float* h2    = ws + o; o += (size_t)NN * 64;    // enc2 out
    float* score = ws + o; o += (size_t)NEDGE;
    float* ex    = ws + o; o += (size_t)NEDGE;
    int*   smax  = (int*)(ws + o); o += (size_t)NN;
    float* denom = ws + o; o += (size_t)NN;
    float* efm   = ws + o; o += 8;
    __bf16* abf  = (__bf16*)(ws + o); o += (size_t)NN * 512 / 2;  // N*512 bf16
    __bf16* w1bf = (__bf16*)(ws + o); o += (512 * 512) / 2;       // weight staging
    __bf16* w2bf = (__bf16*)(ws + o); o += (512 * 512) / 2;
    float* mu    = xl;     // safe reuse: consumed before dec layers rewrite xl/xr
    float* lv    = xr;

    float* z_out = (float*)d_out;             // [N, 64]
    float* r_out = z_out + (size_t)NN * HID2; // [N, 512]

    // Self-loop edge attr = mean of edge_feat.
    hipMemsetAsync(efm, 0, 8 * sizeof(float), stream);
    ef_mean_accum<<<256, 256, 0, stream>>>(ef, efm);
    scale_vec<<<1, 8, 0, stream>>>(efm, 1.0f / (float)EEDGE, 8);

    // Encoder.
    run_gat_layer(x,  D_IN, HID1, eWl0, eWr0, eWe0, eatt0, eb0, ei, ef, efm,
                  abf, w1bf, w2bf, xl, xr, score, ex, smax, denom, h1, stream);
    run_gat_layer(h1, HID1, HID2, eWl1, eWr1, eWe1, eatt1, eb1, ei, ef, efm,
                  abf, w1bf, w2bf, xl, xr, score, ex, smax, denom, h2, stream);

    // mu / logvar (fused dual GEMM over shared h2) + reparameterize.
    {
        const size_t na = (size_t)NN * HID2;
        const size_t nw = (size_t)HID2 * HID2;
        cvt_bf16<<<(na + 255) / 256, 256, 0, stream>>>(h2, abf, na);
        cvt_bf16<<<(nw + 255) / 256, 256, 0, stream>>>(muW, w1bf, nw);
        cvt_bf16<<<(nw + 255) / 256, 256, 0, stream>>>(lvW, w2bf, nw);
        dim3 gblk(32, 4);
        dim3 ggrid(HID2 / 16, (NN / 16 + 3) / 4);
        wmma_gemm_dual<<<ggrid, gblk, 0, stream>>>(abf, w1bf, w2bf, mub, lvb,
                                                   mu, lv, NN, HID2, HID2);
    }
    reparam<<<((size_t)NN * HID2 + 255) / 256, 256, 0, stream>>>(mu, lv, eps, z_out, NN * HID2);

    // Decoder (z lives in d_out; dec layers recompute xl/xr, overwriting mu/lv).
    run_gat_layer(z_out, HID2, HID1, dWl0, dWr0, dWe0, datt0, db0, ei, ef, efm,
                  abf, w1bf, w2bf, xl, xr, score, ex, smax, denom, h1, stream);
    run_gat_layer(h1, HID1, D_IN, dWl1, dWr1, dWe1, datt1, db1, ei, ef, efm,
                  abf, w1bf, w2bf, xl, xr, score, ex, smax, denom, r_out, stream);
}
